// CAR_61143154425900
// MI455X (gfx1250) — compile-verified, matrix operand order checked by reading, not run on hardware
//
#include <hip/hip_runtime.h>
#include <hip/hip_bf16.h>

#define TOK   4096      // 4 * 1024 tokens
#define NBATCH 4
#define LSEQ  1024
#define DM    256
#define DI    512
#define DSTATE 16
#define DTR   16
#define KC    64        // K chunk staged per LDS buffer

typedef __attribute__((ext_vector_type(16))) _Float16 v16h;
typedef __attribute__((ext_vector_type(8)))  _Float16 v8h;
typedef __attribute__((ext_vector_type(8)))  float    v8f;
typedef __attribute__((ext_vector_type(4)))  int      v4i;

#if defined(__AMDGCN__) && __has_builtin(__builtin_amdgcn_global_load_async_to_lds_b128) && __has_builtin(__builtin_amdgcn_s_wait_asynccnt)
#define USE_ASYNC 1
// Builtin signature (from clang diagnostic): param0 = AS1 (global, prints as
// __device__) v4i*, param1 = AS3 (LDS) v4i*, then imm offset + imm cpol.
#define GPTR(p) ((__attribute__((address_space(1))) v4i*)(void*)(p))
#define LPTR(p) ((__attribute__((address_space(3))) v4i*)(void*)(p))
#else
#define USE_ASYNC 0
#endif

// ---------------------------------------------------------------------------
// Fragment load: 16 f16 per lane, ISA 16-bit 16x32 layout.
// lane group 0 (lanes 0-15): K = [0..7] then [16..23]
// lane group 1 (lanes16-31): K = [8..15] then [24..31]
// Caller passes p = row_base + (lane>>4)*8 + k0; we grab p[0..7] and p[16..23].
// Works for both global and LDS pointers (address space inferred at inline).
// ---------------------------------------------------------------------------
__device__ __forceinline__ v16h load_frag16(const _Float16* p) {
  union U { v16h v; struct { v8h lo; v8h hi; } s; } u;
  u.s.lo = *(const v8h*)(p);
  u.s.hi = *(const v8h*)(p + 16);
  return u.v;
}

// ---------------------------------------------------------------------------
// Block WMMA GEMM: C[M,N] = A[M,Kp] * B[N,Kp]^T (+ bias[N])
// 4 waves / block; block tile 64x64 (wave w: rows [by*64+w*16, +16), all 64 n).
// B panel (64 x KC) double-buffered in LDS via async-to-LDS copies; fragments
// re-read with ds_load_b128. A fragments loaded straight from global.
// Requires M%64==0, N%64==0, Kp%KC==0.
// ---------------------------------------------------------------------------
__global__ __launch_bounds__(128) void gemm_wmma_block_kernel(
    const _Float16* __restrict__ A, const _Float16* __restrict__ B,
    const float* __restrict__ bias, float* __restrict__ C,
    int M, int N, int Kp)
{
  __shared__ _Float16 lb[2][64 * KC];     // 2 x 8 KB

  const int tid  = threadIdx.x;
  const int wave = tid >> 5;
  const int lane = tid & 31;
  const int r    = lane & 15;
  const int hi   = lane >> 4;
  const int n0   = blockIdx.x * 64;
  const int m0   = blockIdx.y * 64 + wave * 16;

  // Cooperative B staging: thread -> (row = tid&63, 64-byte segment = tid>>6).
  const int br   = tid & 63;
  const int bseg = tid >> 6;                                  // 0 or 1
  const _Float16* gB = B + (size_t)(n0 + br) * Kp + bseg * 32;

  const _Float16* ap = A + (size_t)(m0 + r) * Kp + hi * 8;

  const v8f vz = {0.f, 0.f, 0.f, 0.f, 0.f, 0.f, 0.f, 0.f};
  v8f acc[4];
#pragma unroll
  for (int t = 0; t < 4; ++t) acc[t] = vz;

  auto stage = [&](int c, int buf) {
    const _Float16* g = gB + (size_t)c * KC;
    _Float16* l = &lb[buf][br * KC + bseg * 32];
#if USE_ASYNC
    __builtin_amdgcn_global_load_async_to_lds_b128(GPTR(g),      LPTR(l),      0, 0);
    __builtin_amdgcn_global_load_async_to_lds_b128(GPTR(g + 8),  LPTR(l + 8),  0, 0);
    __builtin_amdgcn_global_load_async_to_lds_b128(GPTR(g + 16), LPTR(l + 16), 0, 0);
    __builtin_amdgcn_global_load_async_to_lds_b128(GPTR(g + 24), LPTR(l + 24), 0, 0);
#else
    *(v8h*)(l)      = *(const v8h*)(g);
    *(v8h*)(l + 8)  = *(const v8h*)(g + 8);
    *(v8h*)(l + 16) = *(const v8h*)(g + 16);
    *(v8h*)(l + 24) = *(const v8h*)(g + 24);
#endif
  };

  const int nChunks = Kp / KC;
  stage(0, 0);
#if USE_ASYNC
  __builtin_amdgcn_s_wait_asynccnt(0);
#endif
  __syncthreads();

  for (int c = 0; c < nChunks; ++c) {
    const int cur = c & 1;
    if (c + 1 < nChunks) stage(c + 1, cur ^ 1);   // prefetch next panel

#pragma unroll
    for (int kk = 0; kk < KC; kk += 32) {
      v16h av = load_frag16(ap + c * KC + kk);
#pragma unroll
      for (int t = 0; t < 4; ++t) {
        v16h bv = load_frag16(&lb[cur][(t * 16 + r) * KC + hi * 8 + kk]);
        acc[t] = __builtin_amdgcn_wmma_f32_16x16x32_f16(
            false, av, false, bv, (short)0, acc[t], false, false);
      }
    }
#if USE_ASYNC
    __builtin_amdgcn_s_wait_asynccnt(0);
#endif
    __syncthreads();
  }

#pragma unroll
  for (int t = 0; t < 4; ++t) {
    int n = n0 + t * 16 + r;                 // C/D: lane&15 -> column
    float bn = bias ? bias[n] : 0.f;
#pragma unroll
    for (int i = 0; i < 8; ++i)              // vgpr i -> row i + 8*hi
      C[(size_t)(m0 + hi * 8 + i) * N + n] = acc[t][i] + bn;
  }
}

// ---------------------------------------------------------------------------
// Fallback single-wave WMMA GEMM for N not a multiple of 64 (x-proj, N=48).
// ---------------------------------------------------------------------------
template<int NT>
__global__ __launch_bounds__(32) void gemm_wmma_kernel(
    const _Float16* __restrict__ A, const _Float16* __restrict__ B,
    const float* __restrict__ bias, float* __restrict__ C,
    int M, int N, int Kp)
{
  const int lane = threadIdx.x & 31;
  const int r    = lane & 15;
  const int hi   = lane >> 4;
  const int n0   = blockIdx.x * (16 * NT);
  const int m0   = blockIdx.y * 16;

  const _Float16* ap  = A + (size_t)(m0 + r) * Kp + hi * 8;
  const _Float16* bp0 = B + (size_t)(n0 + r) * Kp + hi * 8;

  const v8f vz = {0.f, 0.f, 0.f, 0.f, 0.f, 0.f, 0.f, 0.f};
  v8f acc[NT];
#pragma unroll
  for (int t = 0; t < NT; ++t) acc[t] = vz;

  for (int k0 = 0; k0 < Kp; k0 += 32) {
    v16h av = load_frag16(ap + k0);
#pragma unroll
    for (int t = 0; t < NT; ++t) {
      v16h bv = load_frag16(bp0 + (size_t)t * 16 * Kp + k0);
      acc[t] = __builtin_amdgcn_wmma_f32_16x16x32_f16(
          false, av, false, bv, (short)0, acc[t], false, false);
    }
  }

#pragma unroll
  for (int t = 0; t < NT; ++t) {
    int n = n0 + t * 16 + r;
    float bn = bias ? bias[n] : 0.f;
#pragma unroll
    for (int i = 0; i < 8; ++i)
      C[(size_t)(m0 + hi * 8 + i) * N + n] = acc[t][i] + bn;
  }
}

// ---------------------------------------------------------------------------
// f32 -> f16 staging with K zero-pad and sub-matrix extraction.
// ---------------------------------------------------------------------------
__global__ __launch_bounds__(256) void cvt_f32_f16_kernel(
    const float* __restrict__ src, _Float16* __restrict__ dst,
    int rows, int Kin, int Kp, int srcStride, int srcOff)
{
  int i = blockIdx.x * blockDim.x + threadIdx.x;
  if (i >= rows * Kp) return;
  int rr = i / Kp;
  int k  = i - rr * Kp;
  float v = (k < Kin) ? src[(size_t)rr * srcStride + srcOff + k] : 0.f;
  dst[i] = (_Float16)v;
}

// ---------------------------------------------------------------------------
// Layernorm over 256-wide rows; optional exact-erf GELU. One block per row.
// ---------------------------------------------------------------------------
__global__ __launch_bounds__(256) void ln_kernel(
    const float* __restrict__ x, const float* __restrict__ g,
    const float* __restrict__ b, float* __restrict__ out, int doGelu)
{
  __shared__ float s1[256];
  __shared__ float s2[256];
  const int row = blockIdx.x;
  const int tid = threadIdx.x;
  float v = x[(size_t)row * DM + tid];
  s1[tid] = v;
  s2[tid] = v * v;
  __syncthreads();
  for (int off = 128; off > 0; off >>= 1) {
    if (tid < off) { s1[tid] += s1[tid + off]; s2[tid] += s2[tid + off]; }
    __syncthreads();
  }
  float mu  = s1[0] * (1.f / DM);
  float var = s2[0] * (1.f / DM) - mu * mu;
  float y = (v - mu) / sqrtf(var + 1e-5f) * g[tid] + b[tid];
  if (doGelu) y = 0.5f * y * (1.f + erff(y * 0.70710678118654752f));
  out[(size_t)row * DM + tid] = y;
}

// ---------------------------------------------------------------------------
// Causal depthwise conv (width 4) + SiLU. u = xz[..., :512].
// ---------------------------------------------------------------------------
__global__ __launch_bounds__(256) void conv_silu_kernel(
    const float* __restrict__ xz, const float* __restrict__ cw,
    const float* __restrict__ cb, float* __restrict__ uc)
{
  int idx = blockIdx.x * blockDim.x + threadIdx.x;
  if (idx >= TOK * DI) return;
  int c = idx & (DI - 1);
  int t = idx >> 9;             // token = b*L + l
  int l = t & (LSEQ - 1);
  int b = t >> 10;
  float s = cb[c];
#pragma unroll
  for (int j = 0; j < 4; ++j) {
    int ls = l - 3 + j;
    if (ls >= 0) s += cw[c * 4 + j] * xz[(size_t)(b * LSEQ + ls) * (2 * DI) + c];
  }
  uc[(size_t)t * DI + c] = s / (1.f + expf(-s));
}

__global__ __launch_bounds__(256) void bias_softplus_kernel(
    float* __restrict__ d, const float* __restrict__ bvec)
{
  int i = blockIdx.x * blockDim.x + threadIdx.x;
  if (i >= TOK * DI) return;
  float x = d[i] + bvec[i & (DI - 1)];
  d[i] = (x > 20.f) ? x : log1pf(expf(x));
}

// ---------------------------------------------------------------------------
// Selective scan: one lane per (b, d, state); 16-lane shuffle reduce for y.
// ---------------------------------------------------------------------------
__global__ __launch_bounds__(256) void scan_kernel(
    const float* __restrict__ uc, const float* __restrict__ delta,
    const float* __restrict__ xdbc, const float* __restrict__ Alog,
    const float* __restrict__ Dp, float* __restrict__ y)
{
  int gid  = blockIdx.x * blockDim.x + threadIdx.x;   // NBATCH*DI*DSTATE lanes
  int st   = gid & (DSTATE - 1);
  int pair = gid >> 4;
  int d    = pair & (DI - 1);
  int b    = pair >> 9;
  float Ac = -expf(Alog[d * DSTATE + st]);
  float Dv = Dp[d];
  float s  = 0.f;
  for (int l = 0; l < LSEQ; ++l) {
    size_t t  = (size_t)(b * LSEQ + l);
    float dt  = delta[t * DI + d];
    float u   = uc[t * DI + d];
    float Bm  = xdbc[t * 48 + DTR + st];
    float Cc  = xdbc[t * 48 + DTR + DSTATE + st];
    s = s * expf(dt * Ac) + dt * u * Bm;
    float p = s * Cc;
    p += __shfl_xor(p, 1);
    p += __shfl_xor(p, 2);
    p += __shfl_xor(p, 4);
    p += __shfl_xor(p, 8);
    if (st == 0) y[t * DI + d] = p + u * Dv;
  }
}

__global__ __launch_bounds__(256) void gate_silu_kernel(
    float* __restrict__ y, const float* __restrict__ xz)
{
  int i = blockIdx.x * blockDim.x + threadIdx.x;
  if (i >= TOK * DI) return;
  int c = i & (DI - 1);
  int t = i >> 9;
  float z = xz[(size_t)t * (2 * DI) + DI + c];
  y[i] *= z / (1.f + expf(-z));
}

__global__ __launch_bounds__(256) void add_kernel(
    float* __restrict__ h, const float* __restrict__ t1)
{
  int i = blockIdx.x * blockDim.x + threadIdx.x;
  if (i >= TOK * DM) return;
  h[i] += t1[i];
}

__global__ __launch_bounds__(256) void gelu_kernel(float* __restrict__ h)
{
  int i = blockIdx.x * blockDim.x + threadIdx.x;
  if (i >= TOK * DM) return;
  float v = h[i];
  h[i] = 0.5f * v * (1.f + erff(v * 0.70710678118654752f));
}

// 0.01 * mean over rows of ||row||_2  -> atomicAdd into scalar
__global__ __launch_bounds__(256) void l2_kernel(
    const float* __restrict__ h, float* __restrict__ accum)
{
  __shared__ float s[256];
  const int row = blockIdx.x;
  const int tid = threadIdx.x;
  float v = h[(size_t)row * DM + tid];
  s[tid] = v * v;
  __syncthreads();
  for (int off = 128; off > 0; off >>= 1) {
    if (tid < off) s[tid] += s[tid + off];
    __syncthreads();
  }
  if (tid == 0) atomicAdd(accum, 0.01f * sqrtf(s[0]) * (1.f / TOK));
}

// ---------------------------------------------------------------------------
extern "C" void kernel_launch(void* const* d_in, const int* in_sizes, int n_in,
                              void* d_out, int out_size, void* d_ws, size_t ws_size,
                              hipStream_t stream)
{
  (void)in_sizes; (void)n_in; (void)ws_size;
  const float* x       = (const float*)d_in[0];
  const float* in_w    = (const float*)d_in[1];
  const float* in_b    = (const float*)d_in[2];
  const float* ln_g    = (const float*)d_in[3];
  const float* ln_b    = (const float*)d_in[4];
  const float* blk_ng  = (const float*)d_in[5];
  const float* blk_nb  = (const float*)d_in[6];
  const float* blk_ipw = (const float*)d_in[7];
  const float* blk_cw  = (const float*)d_in[8];
  const float* blk_cb  = (const float*)d_in[9];
  const float* blk_xpw = (const float*)d_in[10];
  const float* blk_dtw = (const float*)d_in[11];
  const float* blk_dtb = (const float*)d_in[12];
  const float* blk_Alog= (const float*)d_in[13];
  const float* blk_D   = (const float*)d_in[14];
  const float* blk_opw = (const float*)d_in[15];
  const float* op_w    = (const float*)d_in[16];
  const float* op_b    = (const float*)d_in[17];
  const float* cls_w   = (const float*)d_in[18];
  const float* cls_b   = (const float*)d_in[19];
  float* out = (float*)d_out;

  unsigned char* bump = (unsigned char*)d_ws;
  auto alloc = [&](size_t bytes) -> void* {
    void* r = (void*)bump;
    bump += (bytes + 255) & ~(size_t)255;
    return r;
  };

  _Float16* A16  = (_Float16*)alloc((size_t)TOK * 768 * sizeof(_Float16));
  _Float16* Win  = (_Float16*)alloc((size_t)DM  * 768 * sizeof(_Float16));
  _Float16* Wip  = (_Float16*)alloc((size_t)2 * 1024 * DM * sizeof(_Float16));
  _Float16* Wxp  = (_Float16*)alloc((size_t)2 * 48 * DI * sizeof(_Float16));
  _Float16* Wdt  = (_Float16*)alloc((size_t)2 * DI * KC * sizeof(_Float16));
  _Float16* Wop  = (_Float16*)alloc((size_t)2 * DM * DI * sizeof(_Float16));
  _Float16* Wopw = (_Float16*)alloc((size_t)DM * DM * sizeof(_Float16));
  _Float16* Wcls = (_Float16*)alloc((size_t)128 * DM * sizeof(_Float16));
  float* H     = (float*)alloc((size_t)TOK * DM * 4);
  float* HN    = (float*)alloc((size_t)TOK * DM * 4);
  float* XZ    = (float*)alloc((size_t)TOK * 2 * DI * 4);
  float* UC    = (float*)alloc((size_t)TOK * DI * 4);
  float* XDBC  = (float*)alloc((size_t)TOK * 48 * 4);
  float* DELTA = (float*)alloc((size_t)TOK * DI * 4);
  float* Y     = (float*)alloc((size_t)TOK * DI * 4);
  float* T1    = (float*)alloc((size_t)TOK * DM * 4);

  auto cvt = [&](const float* src, _Float16* dst, int rows, int Kin, int Kp,
                 int stride, int off) {
    int n = rows * Kp;
    cvt_f32_f16_kernel<<<(n + 255) / 256, 256, 0, stream>>>(src, dst, rows, Kin, Kp, stride, off);
  };
  auto gemm = [&](const _Float16* A, const _Float16* B, const float* bias,
                  float* C, int M, int N, int Kp) {
    if ((N & 63) == 0 && (Kp & (KC - 1)) == 0)
      gemm_wmma_block_kernel<<<dim3(N / 64, M / 64), 128, 0, stream>>>(A, B, bias, C, M, N, Kp);
    else
      gemm_wmma_kernel<1><<<dim3(N / 16, M / 16), 32, 0, stream>>>(A, B, bias, C, M, N, Kp);
  };

  // Stage all weights to f16 (deterministic, re-done each call).
  cvt(in_w, Win, DM, 768, 768, 768, 0);
  for (int l = 0; l < 2; ++l) {
    cvt(blk_ipw + (size_t)l * 1024 * DM, Wip + (size_t)l * 1024 * DM, 1024, DM, DM, DM, 0);
    cvt(blk_xpw + (size_t)l * 48 * DI,   Wxp + (size_t)l * 48 * DI,   48,   DI, DI, DI, 0);
    cvt(blk_dtw + (size_t)l * DI * DTR,  Wdt + (size_t)l * DI * KC,   DI,   DTR, KC, DTR, 0);
    cvt(blk_opw + (size_t)l * DM * DI,   Wop + (size_t)l * DM * DI,   DM,   DI, DI, DI, 0);
  }
  cvt(op_w,  Wopw, DM,  DM, DM, DM, 0);
  cvt(cls_w, Wcls, 128, DM, DM, DM, 0);

  // Stem: h = gelu(layernorm(x @ in_w.T + in_b))
  cvt(x, A16, TOK, 768, 768, 768, 0);
  gemm(A16, Win, in_b, H, TOK, DM, 768);
  ln_kernel<<<TOK, 256, 0, stream>>>(H, ln_g, ln_b, H, 1);

  for (int l = 0; l < 2; ++l) {
    ln_kernel<<<TOK, 256, 0, stream>>>(H, blk_ng + l * DM, blk_nb + l * DM, HN, 0);
    cvt(HN, A16, TOK, DM, DM, DM, 0);
    gemm(A16, Wip + (size_t)l * 1024 * DM, nullptr, XZ, TOK, 2 * DI, DM);
    conv_silu_kernel<<<(TOK * DI + 255) / 256, 256, 0, stream>>>(
        XZ, blk_cw + (size_t)l * DI * 4, blk_cb + (size_t)l * DI, UC);
    cvt(UC, A16, TOK, DI, DI, DI, 0);
    gemm(A16, Wxp + (size_t)l * 48 * DI, nullptr, XDBC, TOK, 48, DI);
    cvt(XDBC, A16, TOK, DTR, KC, 48, 0);                 // dt cols, K padded to 64
    gemm(A16, Wdt + (size_t)l * DI * KC, nullptr, DELTA, TOK, DI, KC);
    bias_softplus_kernel<<<(TOK * DI + 255) / 256, 256, 0, stream>>>(
        DELTA, blk_dtb + (size_t)l * DI);
    scan_kernel<<<(NBATCH * DI * DSTATE) / 256, 256, 0, stream>>>(
        UC, DELTA, XDBC, blk_Alog + (size_t)l * DI * DSTATE, blk_D + (size_t)l * DI, Y);
    gate_silu_kernel<<<(TOK * DI + 255) / 256, 256, 0, stream>>>(Y, XZ);
    cvt(Y, A16, TOK, DI, DI, DI, 0);
    gemm(A16, Wop + (size_t)l * DM * DI, nullptr, T1, TOK, DM, DI);
    add_kernel<<<(TOK * DM + 255) / 256, 256, 0, stream>>>(H, T1);
  }

  // Head: h = gelu(H @ op_w.T + op_b); logits = h @ cls_w.T + cls_b; l2 loss.
  cvt(H, A16, TOK, DM, DM, DM, 0);
  gemm(A16, Wopw, op_b, T1, TOK, DM, DM);
  gelu_kernel<<<(TOK * DM + 255) / 256, 256, 0, stream>>>(T1);
  cvt(T1, A16, TOK, DM, DM, DM, 0);
  gemm(A16, Wcls, cls_b, out, TOK, 128, DM);
  (void)hipMemsetAsync(out + (out_size - 1), 0, sizeof(float), stream);
  l2_kernel<<<TOK, 256, 0, stream>>>(T1, out + (out_size - 1));
}
